// LSTMGestureModel_54752243089469
// MI455X (gfx1250) — compile-verified, matrix operand order checked by reading, not run on hardware
//
#include <hip/hip_runtime.h>
#include <hip/hip_bf16.h>
#include <math.h>

// Problem constants (match reference)
#define B_   2048
#define T_   256
#define INP  42
#define HID  64
#define G4   256          // 4*HID gate width
#define NOUT 5
#define NBT  (B_/16)      // 128 batch tiles of 16 rows

typedef _Float16 v16h __attribute__((ext_vector_type(16)));
typedef _Float16 v8h  __attribute__((ext_vector_type(8)));
typedef float    v8f  __attribute__((ext_vector_type(8)));

union AF { v16h v; v8h h[2]; };

static __device__ __forceinline__ v8f wmma_f16(v16h a, v16h b, v8f c) {
    // D = A(16x32,f16) x B(32x16,f16) + C(16x16,f32)
    return __builtin_amdgcn_wmma_f32_16x16x32_f16(false, a, false, b, (short)0, c, false, false);
}
static __device__ __forceinline__ float fast_sigmoid(float x) {
    return __builtin_amdgcn_rcpf(1.0f + __expf(-x));
}
static __device__ __forceinline__ float fast_tanh(float x) {
    float e = __expf(-2.0f * x);                 // tanh = 2/(1+e^-2x) - 1
    return 2.0f * __builtin_amdgcn_rcpf(1.0f + e) - 1.0f;
}

// A-fragment K mapping (ISA 7.12.2, 16-bit A 16x32):
//   lanes 0-15 : halves 0-7 -> K 0-7,  halves 8-15 -> K 16-23
//   lanes 16-31: halves 0-7 -> K 8-15, halves 8-15 -> K 24-31
static __device__ __forceinline__ int a_kmap(int j /*0..31*/, int lane) {
    int j2 = j & 15;
    int kl = j2 + ((j2 >= 8) ? 8 : 0) + ((lane >= 16) ? 8 : 0);
    return kl + ((j >= 16) ? 32 : 0);            // second v16h covers K 32..63
}

// ---------------------------------------------------------------------------
// Pack x [B][T][42] f32 -> packed A-fragment layout f16:
//   frag (t,bt) at frag*1024 halves; per lane 32 contiguous halves (a0|a1).
__global__ void k_pack_x(const float* __restrict__ x, _Float16* __restrict__ xp) {
    size_t tid = (size_t)blockIdx.x * blockDim.x + threadIdx.x;  // over B*T*64 halves
    int    j    = (int)(tid & 31);
    int    lane = (int)((tid >> 5) & 31);
    size_t frag = tid >> 10;
    int    t    = (int)(frag >> 7);        // frag / NBT  (NBT==128)
    int    bt   = (int)(frag & (NBT - 1));
    int    m    = lane & 15;
    int    k    = a_kmap(j, lane);
    int    b    = bt * 16 + m;
    float v = (k < INP) ? x[((size_t)b * T_ + t) * INP + k] : 0.0f;
    xp[tid] = (_Float16)v;
}

// ---------------------------------------------------------------------------
// Pack W [G4][Kin] f32 -> f16 WMMA B fragments, per-lane contiguous 16 halves.
// Frag fidx = ntile*2 + kstep.  B[k][n] = W[n][k]  (WMMA computes h @ W^T).
__global__ void k_pack_w(const float* __restrict__ W, _Float16* __restrict__ wp, int Kin) {
    int tid  = blockIdx.x * blockDim.x + threadIdx.x;  // 32 frags * 512 = 16384
    int j    = tid & 15;
    int lane = (tid >> 4) & 31;
    int fidx = tid >> 9;
    int kstep = fidx & 1, ntile = fidx >> 1;
    int klocal = j + ((j >= 8) ? 8 : 0) + ((lane >= 16) ? 8 : 0);
    int k = kstep * 32 + klocal;
    int n = ntile * 16 + (lane & 15);
    float v = (k < Kin) ? W[n * Kin + k] : 0.0f;
    wp[tid] = (_Float16)v;   // wp index == tid by construction
}

// Combined gate biases (b_ih + b_hh) for both layers
__global__ void k_bias(const float* __restrict__ bi0, const float* __restrict__ bh0,
                       const float* __restrict__ bi1, const float* __restrict__ bh1,
                       float* __restrict__ bc0, float* __restrict__ bc1) {
    int i = blockIdx.x * blockDim.x + threadIdx.x;
    if (i < G4)            bc0[i] = bi0[i] + bh0[i];
    else if (i < 2 * G4)   bc1[i - G4] = bi1[i - G4] + bh1[i - G4];
}

// ---------------------------------------------------------------------------
// Input projection: gx = X @ W_ih^T + (b_ih + b_hh), all B*T tokens parallel.
// Input X is in packed A-fragment layout (64B contiguous per lane per frag).
// Output in WMMA C-fragment layout:
//   frag (t,bt,w,g) base = (((t*NBT+bt)*4 + w)*4 + g)*256 floats, +lane*8
__global__ __launch_bounds__(128) void k_proj(const _Float16* __restrict__ xp,  // packed A frags
                                              const _Float16* __restrict__ wp,  // 32 B-frags
                                              const float*    __restrict__ bc,  // [256] bias
                                              float*          __restrict__ gx)
{
    int blk = blockIdx.x;                  // t*NBT + bt
    int w = threadIdx.x >> 5;
    int L = threadIdx.x & 31;

    const _Float16* abase = xp + (size_t)blk * 1024 + (size_t)L * 32;
    v16h a0 = *(const v16h*)(abase);
    v16h a1 = *(const v16h*)(abase + 16);

    size_t gbase = (((size_t)blk * 4 + w) * 4) * 256 + (size_t)L * 8;
    #pragma unroll
    for (int g = 0; g < 4; ++g) {
        int nt = g * 4 + w;
        v16h bf0 = *(const v16h*)(wp + (size_t)(nt * 2 + 0) * 512 + L * 16);
        v16h bf1 = *(const v16h*)(wp + (size_t)(nt * 2 + 1) * 512 + L * 16);
        float bias = bc[g * 64 + w * 16 + (L & 15)];   // N = lane&15 for all 8 C elems
        v8f acc = { bias, bias, bias, bias, bias, bias, bias, bias };
        acc = wmma_f16(a0, bf0, acc);
        acc = wmma_f16(a1, bf1, acc);
        *(v8f*)(gx + gbase + (size_t)g * 256) = acc;
    }
}

// ---------------------------------------------------------------------------
// Recurrent layer. One block per 16-batch tile, 4 waves; wave w owns hidden
// columns [16w,16w+16). W_hh fragments resident in VGPRs for all 256 steps.
// h ping-pongs through LDS (1 barrier/step); c stays in accumulator layout.
// gx fragments are software-pipelined one step ahead + L2-prefetched 8 ahead.
// Layer0 (WRITE_ALL): h written to global in packed A-fragment layout, reusing
// the A-fragments built from LDS (2x32B coalesced stores per lane per step).
template<bool WRITE_ALL>
__global__ __launch_bounds__(128) void k_recur(const _Float16* __restrict__ whp,  // W_hh frags
                                               const float*    __restrict__ gx,   // frag layout
                                               _Float16*       __restrict__ hall, // packed A frags (layer0)
                                               float*          __restrict__ hlast)// [B][64]   (layer1)
{
    __shared__ _Float16 hbuf[2][16 * 64];
    int bt = blockIdx.x;
    int w  = threadIdx.x >> 5;
    int L  = threadIdx.x & 31;
    int m  = L & 15;
    int k0 = (L < 16) ? 0 : 8;
    int b0 = bt * 16;
    int rowadd = (L >= 16) ? 8 : 0;        // C/D layout: M = r + 8*(lane>=16)
    int colg   = w * 16 + (L & 15);        // N = lane&15 within wave slice

    // Resident W_hh B-fragments: 4 gates x 2 K-steps (64 VGPRs)
    v16h bw[4][2];
    #pragma unroll
    for (int g = 0; g < 4; ++g)
        #pragma unroll
        for (int ks = 0; ks < 2; ++ks)
            bw[g][ks] = *(const v16h*)(whp + (size_t)((g * 4 + w) * 2 + ks) * 512 + L * 16);

    for (int i = threadIdx.x; i < 2 * 16 * 64; i += 128) hbuf[0][i] = (_Float16)0.0f;
    v8f c = {0.f,0.f,0.f,0.f,0.f,0.f,0.f,0.f};
    __syncthreads();

    const size_t gstep = (size_t)NBT * 16 * 256;                      // floats per t
    const float* gb = gx + (((size_t)bt * 4 + w) * 4) * 256 + (size_t)L * 8;

    // Preload step-0 gate inputs
    v8f ai = *(const v8f*)(gb + 0 * 256);
    v8f af = *(const v8f*)(gb + 1 * 256);
    v8f ag = *(const v8f*)(gb + 2 * 256);
    v8f ao = *(const v8f*)(gb + 3 * 256);

    #pragma unroll 2
    for (int t = 0; t < T_; ++t) {
        int cur = t & 1;
        // Warm L2 ahead of the serial chain
        if (t + 8 < T_) {
            const float* pf = gb + (size_t)(t + 8) * gstep;
            __builtin_prefetch(pf + 0 * 256, 0, 1);
            __builtin_prefetch(pf + 1 * 256, 0, 1);
            __builtin_prefetch(pf + 2 * 256, 0, 1);
            __builtin_prefetch(pf + 3 * 256, 0, 1);
        }

        // A fragments of h(t-1) from LDS (row-major 16x64 tile)
        const _Float16* hrow = hbuf[cur] + m * 64;
        AF a0, a1;
        a0.h[0] = *(const v8h*)(hrow + k0);
        a0.h[1] = *(const v8h*)(hrow + k0 + 16);
        a1.h[0] = *(const v8h*)(hrow + k0 + 32);
        a1.h[1] = *(const v8h*)(hrow + k0 + 48);

        // Layer0: persist h(t-1) in packed A-fragment layout for layer1's proj
        if (WRITE_ALL && t > 0) {
            _Float16* dst = hall + (size_t)((t - 1) * NBT + bt) * 1024 + (size_t)L * 32;
            *(v16h*)(dst)      = a0.v;
            *(v16h*)(dst + 16) = a1.v;
        }

        ai = wmma_f16(a0.v, bw[0][0], ai);  ai = wmma_f16(a1.v, bw[0][1], ai);
        af = wmma_f16(a0.v, bw[1][0], af);  af = wmma_f16(a1.v, bw[1][1], af);
        ag = wmma_f16(a0.v, bw[2][0], ag);  ag = wmma_f16(a1.v, bw[2][1], ag);
        ao = wmma_f16(a0.v, bw[3][0], ao);  ao = wmma_f16(a1.v, bw[3][1], ao);

        // Gate math (f32)
        v8f hv;
        #pragma unroll
        for (int r = 0; r < 8; ++r) {
            float iv = fast_sigmoid(ai[r]);
            float fv = fast_sigmoid(af[r]);
            float gv = fast_tanh(ag[r]);
            float ov = fast_sigmoid(ao[r]);
            float cv = fv * c[r] + iv * gv;
            c[r] = cv;
            hv[r] = ov * fast_tanh(cv);
        }

        // Software-pipeline next step's gate inputs (off the critical path)
        {
            int tn = (t + 1 < T_) ? (t + 1) : t;
            const float* gn = gb + (size_t)tn * gstep;
            ai = *(const v8f*)(gn + 0 * 256);
            af = *(const v8f*)(gn + 1 * 256);
            ag = *(const v8f*)(gn + 2 * 256);
            ao = *(const v8f*)(gn + 3 * 256);
        }

        // Store h(t) into the other LDS buffer (ping-pong -> 1 barrier/step)
        _Float16* hw = hbuf[cur ^ 1];
        #pragma unroll
        for (int r = 0; r < 8; ++r)
            hw[(r + rowadd) * 64 + colg] = (_Float16)hv[r];

        if (!WRITE_ALL && t == T_ - 1) {
            #pragma unroll
            for (int r = 0; r < 8; ++r)
                hlast[(size_t)(b0 + r + rowadd) * HID + colg] = hv[r];
        }
        __syncthreads();
    }

    // Epilogue (layer0): persist h(T-1) fragment
    if (WRITE_ALL) {
        const _Float16* hrow = hbuf[T_ & 1] + m * 64;
        AF a0, a1;
        a0.h[0] = *(const v8h*)(hrow + k0);
        a0.h[1] = *(const v8h*)(hrow + k0 + 16);
        a1.h[0] = *(const v8h*)(hrow + k0 + 32);
        a1.h[1] = *(const v8h*)(hrow + k0 + 48);
        _Float16* dst = hall + (size_t)((T_ - 1) * NBT + bt) * 1024 + (size_t)L * 32;
        *(v16h*)(dst)      = a0.v;
        *(v16h*)(dst + 16) = a1.v;
    }
}

// ---------------------------------------------------------------------------
// Head: logits = h_last @ W_fc^T + b_fc; softmax over 5 classes.
__global__ __launch_bounds__(256) void k_head(const float* __restrict__ hlast,
                                              const float* __restrict__ Wfc,
                                              const float* __restrict__ bfc,
                                              float* __restrict__ out)
{
    __shared__ float sW[NOUT * HID];
    __shared__ float sb[NOUT];
    for (int i = threadIdx.x; i < NOUT * HID; i += blockDim.x) sW[i] = Wfc[i];
    if (threadIdx.x < NOUT) sb[threadIdx.x] = bfc[threadIdx.x];
    __syncthreads();

    int b = blockIdx.x * blockDim.x + threadIdx.x;
    float logits[NOUT];
    #pragma unroll
    for (int o = 0; o < NOUT; ++o) {
        float acc = sb[o];
        #pragma unroll 8
        for (int k = 0; k < HID; ++k) acc += hlast[(size_t)b * HID + k] * sW[o * HID + k];
        logits[o] = acc;
    }
    float mx = logits[0];
    #pragma unroll
    for (int o = 1; o < NOUT; ++o) mx = fmaxf(mx, logits[o]);
    float e[NOUT], s = 0.0f;
    #pragma unroll
    for (int o = 0; o < NOUT; ++o) { e[o] = __expf(logits[o] - mx); s += e[o]; }
    float rs = 1.0f / s;
    #pragma unroll
    for (int o = 0; o < NOUT; ++o) out[(size_t)b * NOUT + o] = e[o] * rs;
}

// ---------------------------------------------------------------------------
extern "C" void kernel_launch(void* const* d_in, const int* in_sizes, int n_in,
                              void* d_out, int out_size, void* d_ws, size_t ws_size,
                              hipStream_t stream)
{
    const float* x    = (const float*)d_in[0];
    const float* Wih0 = (const float*)d_in[1];
    const float* Whh0 = (const float*)d_in[2];
    const float* bih0 = (const float*)d_in[3];
    const float* bhh0 = (const float*)d_in[4];
    const float* Wih1 = (const float*)d_in[5];
    const float* Whh1 = (const float*)d_in[6];
    const float* bih1 = (const float*)d_in[7];
    const float* bhh1 = (const float*)d_in[8];
    const float* Wfc  = (const float*)d_in[9];
    const float* bfc  = (const float*)d_in[10];

    char* ws = (char*)d_ws;
    size_t off = 0;
    auto alloc = [&](size_t bytes) -> char* {
        char* p = ws + off;
        off += (bytes + 255) & ~(size_t)255;
        return p;
    };
    _Float16* xp    = (_Float16*)alloc((size_t)B_ * T_ * 64 * 2);        // 64 MB, packed A frags
    _Float16* h0p   = (_Float16*)alloc((size_t)B_ * T_ * 64 * 2);        // 64 MB, packed A frags
    float*    gx    = (float*)   alloc((size_t)T_ * NBT * 16 * 256 * 4); // 512 MB (reused per layer)
    _Float16* wpih0 = (_Float16*)alloc(32 * 512 * 2);
    _Float16* wphh0 = (_Float16*)alloc(32 * 512 * 2);
    _Float16* wpih1 = (_Float16*)alloc(32 * 512 * 2);
    _Float16* wphh1 = (_Float16*)alloc(32 * 512 * 2);
    float*    bc0   = (float*)alloc(G4 * 4);
    float*    bc1   = (float*)alloc(G4 * 4);
    float*    hlast = (float*)alloc((size_t)B_ * HID * 4);

    k_pack_x<<<(B_ * T_ * 64) / 256, 256, 0, stream>>>(x, xp);
    k_pack_w<<<64, 256, 0, stream>>>(Wih0, wpih0, INP);
    k_pack_w<<<64, 256, 0, stream>>>(Whh0, wphh0, HID);
    k_pack_w<<<64, 256, 0, stream>>>(Wih1, wpih1, HID);
    k_pack_w<<<64, 256, 0, stream>>>(Whh1, wphh1, HID);
    k_bias<<<2, 256, 0, stream>>>(bih0, bhh0, bih1, bhh1, bc0, bc1);

    // Layer 0
    k_proj<<<T_ * NBT, 128, 0, stream>>>(xp, wpih0, bc0, gx);
    k_recur<true><<<NBT, 128, 0, stream>>>(wphh0, gx, h0p, nullptr);
    // Layer 1
    k_proj<<<T_ * NBT, 128, 0, stream>>>(h0p, wpih1, bc1, gx);
    k_recur<false><<<NBT, 128, 0, stream>>>(wphh1, gx, nullptr, hlast);
    // Head
    k_head<<<B_ / 256, 256, 0, stream>>>(hlast, Wfc, bfc, (float*)d_out);
}
